// CausalSelfAttention_37400575213966
// MI455X (gfx1250) — matrix operand order repaired and checked by reading, not compiled, not run on hardware
//
#include <hip/hip_runtime.h>
#include <hip/hip_bf16.h>

#define D_K   768
#define T_SEQ 4096
#define BATCH 4
#define NKC   (D_K / 32)   // 24 k-chunks of 32 for 16x16x32 WMMA

typedef __attribute__((ext_vector_type(16))) __bf16        v16bf;
typedef __attribute__((ext_vector_type(2)))  __bf16        bf16x2;
typedef __attribute__((ext_vector_type(8)))  float         v8f;
typedef __attribute__((ext_vector_type(4)))  unsigned int  u32x4;
typedef __attribute__((ext_vector_type(2)))  unsigned int  u32x2;
typedef __attribute__((ext_vector_type(4)))  float         f32x4;
typedef __attribute__((ext_vector_type(2)))  float         f32x2;

union BF16x16 { unsigned int u[8]; u32x4 u4[2]; v16bf v; };

#define WMMA_BF16(A, B, C) \
  __builtin_amdgcn_wmma_f32_16x16x32_bf16(false, (A).v, false, (B).v, (short)0, (C), false, false)

// f32 pair -> packed bf16 dword via a single v_cvt_pk_bf16_f32
__device__ __forceinline__ unsigned int pack_bf16(float a, float b) {
  f32x2 t; t.x = a; t.y = b;
  bf16x2 c = __builtin_convertvector(t, bf16x2);
  return __builtin_bit_cast(unsigned int, c);
}
__device__ __forceinline__ unsigned short f2bf(float a) {
  return __builtin_bit_cast(unsigned short, (__bf16)a);
}

// reductions across the 16-lane halves of a wave32 (rows of a WMMA C tile)
__device__ __forceinline__ float half16_max(float v) {
  v = fmaxf(v, __shfl_xor(v, 1, 32));
  v = fmaxf(v, __shfl_xor(v, 2, 32));
  v = fmaxf(v, __shfl_xor(v, 4, 32));
  v = fmaxf(v, __shfl_xor(v, 8, 32));
  return v;
}
__device__ __forceinline__ float half16_sum(float v) {
  v += __shfl_xor(v, 1, 32);
  v += __shfl_xor(v, 2, 32);
  v += __shfl_xor(v, 4, 32);
  v += __shfl_xor(v, 8, 32);
  return v;
}

// ---------------------------------------------------------------------------
// Kernel 0: one-shot f32 -> bf16 convert (x, Wq, Wk, Wv). 8 floats/thread.
// ---------------------------------------------------------------------------
__global__ __launch_bounds__(256) void cvt_bf16_kernel(
    const float* __restrict__ src, unsigned int* __restrict__ dst, int ngroups)
{
  const int i = blockIdx.x * 256 + threadIdx.x;
  if (i >= ngroups) return;
  const f32x4* s = (const f32x4*)src + 2 * (size_t)i;
  f32x4 p0 = s[0], p1 = s[1];
  u32x4 d;
  d.x = pack_bf16(p0.x, p0.y); d.y = pack_bf16(p0.z, p0.w);
  d.z = pack_bf16(p1.x, p1.y); d.w = pack_bf16(p1.z, p1.w);
  ((u32x4*)dst)[i] = d;
}

// ---------------------------------------------------------------------------
// Kernel 1: pure bf16 GEMM: Q = (x Wq^T)*scale, K = x Wk^T, V^T = (x Wv^T)^T.
// grid = (BT/32, 6), block = 256 (8 waves). Each wave: a 32x16 tile of
// Q, K and V (2 M-tiles x 3 outputs = 6 WMMAs per k-chunk, W B-frags
// amortized over both M-tiles). Fragment loads rotate one iteration ahead
// so global_load_b128s overlap the WMMAs.
// ---------------------------------------------------------------------------
__global__ __launch_bounds__(256) void qkv_proj_kernel(
    const unsigned short* __restrict__ xb,
    const unsigned short* __restrict__ Wqb,
    const unsigned short* __restrict__ Wkb,
    const unsigned short* __restrict__ Wvb,
    unsigned short* __restrict__ Qb, unsigned short* __restrict__ Kb,
    unsigned short* __restrict__ Vtb)
{
  const int g0   = blockIdx.x << 5;               // 32-row tile of [B*T]
  const int lane = threadIdx.x & 31;
  const int wave = threadIdx.x >> 5;
  const int e0   = (blockIdx.y * 8 + wave) << 4;  // output-feature base
  const int mm   = lane & 15;
  const int hi   = lane >> 4;

  const unsigned short* xr0 = xb  + (size_t)(g0 + mm) * D_K;
  const unsigned short* xr1 = xr0 + (size_t)16 * D_K;
  const unsigned short* wq  = Wqb + (size_t)(e0 + mm) * D_K;
  const unsigned short* wk  = Wkb + (size_t)(e0 + mm) * D_K;
  const unsigned short* wv  = Wvb + (size_t)(e0 + mm) * D_K;

  auto loadA = [&](const unsigned short* p, int K0, BF16x16& f) {
    f.u4[0] = *(const u32x4*)(p + K0 + (hi << 3));
    f.u4[1] = *(const u32x4*)(p + K0 + 16 + (hi << 3));
  };
  auto loadB = [&](const unsigned short* p, int K0, BF16x16& f) {
    f.u4[0] = *(const u32x4*)(p + K0 + (hi << 3));
    f.u4[1] = *(const u32x4*)(p + K0 + 16 + (hi << 3));
  };

  v8f cq0 = {}, ck0 = {}, cv0 = {}, cq1 = {}, ck1 = {}, cv1 = {};

  BF16x16 a0, a1, bq, bk, bv;
  loadA(xr0, 0, a0); loadA(xr1, 0, a1);
  loadB(wq, 0, bq);  loadB(wk, 0, bk);  loadB(wv, 0, bv);

  for (int kc = 1; kc <= NKC; ++kc) {
    BF16x16 na0, na1, nbq, nbk, nbv;
    if (kc < NKC) {                       // issue next iteration's loads first
      const int K0 = kc << 5;
      loadA(xr0, K0, na0); loadA(xr1, K0, na1);
      loadB(wq, K0, nbq);  loadB(wk, K0, nbk);  loadB(wv, K0, nbv);
    }
    cq0 = WMMA_BF16(a0, bq, cq0);  cq1 = WMMA_BF16(a1, bq, cq1);
    ck0 = WMMA_BF16(a0, bk, ck0);  ck1 = WMMA_BF16(a1, bk, ck1);
    cv0 = WMMA_BF16(a0, bv, cv0);  cv1 = WMMA_BF16(a1, bv, cv1);
    if (kc < NKC) { a0 = na0; a1 = na1; bq = nbq; bk = nbk; bv = nbv; }
  }

  const float SC = 0.03608439182435161f;   // 1/sqrt(768) folded into Q
  const int b  = g0 >> 12;
  const int t0 = g0 & (T_SEQ - 1);
#pragma unroll
  for (int r = 0; r < 8; ++r) {
    const int row0 = g0 + r + (hi << 3);
    Qb[(size_t)row0 * D_K + e0 + mm] = f2bf(cq0[r] * SC);
    Kb[(size_t)row0 * D_K + e0 + mm] = f2bf(ck0[r]);
    Qb[(size_t)(row0 + 16) * D_K + e0 + mm] = f2bf(cq1[r] * SC);
    Kb[(size_t)(row0 + 16) * D_K + e0 + mm] = f2bf(ck1[r]);
  }
  // V^T: each M-tile's 8 results are time-consecutive -> b128 stores
  unsigned short* vtrow = Vtb + (size_t)(b * D_K + e0 + mm) * T_SEQ + t0 + (hi << 3);
  u32x4 vv;
  vv.x = pack_bf16(cv0[0], cv0[1]); vv.y = pack_bf16(cv0[2], cv0[3]);
  vv.z = pack_bf16(cv0[4], cv0[5]); vv.w = pack_bf16(cv0[6], cv0[7]);
  *(u32x4*)vtrow = vv;
  vv.x = pack_bf16(cv1[0], cv1[1]); vv.y = pack_bf16(cv1[2], cv1[3]);
  vv.z = pack_bf16(cv1[4], cv1[5]); vv.w = pack_bf16(cv1[6], cv1[7]);
  *(u32x4*)(vtrow + 16) = vv;
}

// ---------------------------------------------------------------------------
// Kernel 2: causal flash attention. One wave owns (16-query tile, 384-feature
// half). O accumulator = 24 x v8f = 192 VGPRs. Online softmax deferred to
// once per 32-key chunk; S subtiles share Q A-frags; fragment loads rotate
// one iteration ahead. P transposed via 1KB/wave LDS bounce + s_wait_dscnt.
// ---------------------------------------------------------------------------
__global__ __launch_bounds__(256) void attn_kernel(
    const unsigned short* __restrict__ Qb,
    const unsigned short* __restrict__ Kb,
    const unsigned short* __restrict__ Vtb,
    float* __restrict__ out)
{
  __shared__ unsigned int ldsP[8][16 * 16];   // per-wave 16x32 bf16 P tile (8KB)

  const int tid  = threadIdx.x;
  const int lane = tid & 31;
  const int wave = tid >> 5;
  const int w    = blockIdx.x * 8 + wave;     // 0..2047
  const int qt   = w >> 1;                    // query tile 0..1023
  const int half = w & 1;                     // feature half (2 x 384)
  const int b    = qt >> 8;
  const int q0   = (qt & 255) << 4;
  const int mm   = lane & 15;
  const int hi   = lane >> 4;

  const unsigned short* Qrow  = Qb  + (size_t)(b * T_SEQ + q0 + mm) * D_K;
  const unsigned short* Kbase = Kb  + (size_t)(b * T_SEQ) * D_K;
  const unsigned short* Vbase = Vtb + (size_t)(b * D_K) * T_SEQ;

  auto loadA = [&](const unsigned short* p, int K0, BF16x16& f) {
    f.u4[0] = *(const u32x4*)(p + K0 + (hi << 3));
    f.u4[1] = *(const u32x4*)(p + K0 + 16 + (hi << 3));
  };
  auto loadB = [&](const unsigned short* p, int K0, BF16x16& f) {
    f.u4[0] = *(const u32x4*)(p + K0 + (hi << 3));
    f.u4[1] = *(const u32x4*)(p + K0 + 16 + (hi << 3));
  };

  v8f acc[24];
#pragma unroll
  for (int i = 0; i < 24; ++i) acc[i] = (v8f){};
  float M[8], L[8];
#pragma unroll
  for (int r = 0; r < 8; ++r) { M[r] = -1e30f; L[r] = 0.0f; }

  const int nkeys   = q0 + 16;                // causal: keys 0..q0+15
  const int nchunks = (nkeys + 31) >> 5;
  unsigned short* pS = (unsigned short*)&ldsP[wave][0];

  for (int c = 0; c < nchunks; ++c) {
    const int  kbase = c << 5;
    const bool has2  = (kbase + 16) < nkeys;  // second subtile has live keys?

    // prefetch next chunk's K rows into cache
    if (kbase + 32 < T_SEQ)
      __builtin_prefetch(Kbase + (size_t)(kbase + 32 + mm) * D_K, 0, 1);

    // S = Q(16x768) . K(32 keys x768)^T : 24 (or 48) WMMAs, rotated loads
    v8f s0 = {}; v8f s1 = {};
    const unsigned short* kr0 = Kbase + (size_t)(kbase + mm) * D_K;
    const unsigned short* kr1 = kr0 + (size_t)16 * D_K;
    {
      BF16x16 a, b0, b1;
      loadA(Qrow, 0, a); loadB(kr0, 0, b0);
      if (has2) loadB(kr1, 0, b1);
      for (int kc = 1; kc <= NKC; ++kc) {
        BF16x16 na, nb0, nb1;
        if (kc < NKC) {
          const int K0 = kc << 5;
          loadA(Qrow, K0, na); loadB(kr0, K0, nb0);
          if (has2) loadB(kr1, K0, nb1);
        }
        s0 = WMMA_BF16(a, b0, s0);
        if (has2) s1 = WMMA_BF16(a, b1, s1);
        if (kc < NKC) { a = na; b0 = nb0; if (has2) b1 = nb1; }
      }
    }

    // online softmax, one rescale per 32-key chunk (Q pre-scaled by 1/sqrt(D))
    const int kn0 = kbase + mm;
    const int kn1 = kbase + 16 + mm;
#pragma unroll
    for (int r = 0; r < 8; ++r) {
      const int qm = q0 + r + (hi << 3);
      float sv0 = (kn0 <= qm) ? s0[r] : -1e30f;
      float sv1 = (kn1 <= qm) ? s1[r] : -1e30f;
      float rmax  = half16_max(fmaxf(sv0, sv1));
      float newM  = fmaxf(M[r], rmax);
      float alpha = __expf(M[r] - newM);
      float p0    = __expf(sv0 - newM);
      float p1    = __expf(sv1 - newM);
      L[r] = L[r] * alpha + half16_sum(p0 + p1);
      M[r] = newM;
#pragma unroll
      for (int ft = 0; ft < 24; ++ft) acc[ft][r] *= alpha;
      pS[(r + (hi << 3)) * 32 + mm]      = f2bf(p0);
      pS[(r + (hi << 3)) * 32 + 16 + mm] = f2bf(p1);
    }

    // C-layout -> A-fragment layout for P via LDS (same-wave RAW; wait DS cnt)
    asm volatile("s_wait_dscnt 0" ::: "memory");
    BF16x16 ap;
    {
      const unsigned int* lp = &ldsP[wave][mm * 16 + (hi << 2)];
      ap.u4[0] = *(const u32x4*)lp;
      ap.u4[1] = *(const u32x4*)(lp + 8);
    }

    // O += P(16x32) . V(32 keys x 384 feats), V stored [D,T]; rotated loads
    {
      const unsigned short* vr0 = Vbase + (size_t)(half * 384 + mm) * T_SEQ + kbase;
      BF16x16 bv;
      loadB(vr0, 0, bv);   // K0 here offsets along T, reuse loadB shape (16 keys)
      // NOTE: loadB's K0 is an element offset; V rows are along T so pass 0 and
      // advance the pointer per feature tile instead.
#pragma unroll
      for (int ft = 0; ft < 24; ++ft) {
        BF16x16 nbv;
        if (ft < 23) {
          const unsigned short* vrn = vr0 + (size_t)((ft + 1) << 4) * T_SEQ;
          loadB(vrn, 0, nbv);
        }
        acc[ft] = WMMA_BF16(ap, bv, acc[ft]);
        if (ft < 23) bv = nbv;
      }
    }
  }

  // epilogue: O / L, fp32 store (lane-adjacent columns -> coalesced)
#pragma unroll
  for (int r = 0; r < 8; ++r) {
    const float inv = 1.0f / L[r];
    const int qrow  = q0 + r + (hi << 3);
    float* orow = out + (size_t)(b * T_SEQ + qrow) * D_K + half * 384;
#pragma unroll
    for (int ft = 0; ft < 24; ++ft) orow[(ft << 4) + mm] = acc[ft][r] * inv;
  }
}

extern "C" void kernel_launch(void* const* d_in, const int* in_sizes, int n_in,
                              void* d_out, int out_size, void* d_ws, size_t ws_size,
                              hipStream_t stream) {
  const float* x  = (const float*)d_in[0];
  const float* Wq = (const float*)d_in[1];
  const float* Wk = (const float*)d_in[2];
  const float* Wv = (const float*)d_in[3];

  const size_t elems  = (size_t)BATCH * T_SEQ * D_K;  // 12.58M bf16 each
  const size_t welems = (size_t)D_K * D_K;
  unsigned short* Qb  = (unsigned short*)d_ws;
  unsigned short* Kb  = Qb + elems;
  unsigned short* Vtb = Kb + elems;                   // [B, D, T] transposed V
  unsigned short* xbb = Vtb + elems;                  // bf16 copy of x
  unsigned short* Wqb = xbb + elems;
  unsigned short* Wkb = Wqb + welems;
  unsigned short* Wvb = Wkb + welems;

  // one-shot f32 -> bf16 converts (bandwidth-bound)
  {
    const int xg = (int)(elems / 8);
    const int wg = (int)(welems / 8);
    cvt_bf16_kernel<<<dim3((xg + 255) / 256), dim3(256), 0, stream>>>(x,  (unsigned int*)xbb, xg);
    cvt_bf16_kernel<<<dim3((wg + 255) / 256), dim3(256), 0, stream>>>(Wq, (unsigned int*)Wqb, wg);
    cvt_bf16_kernel<<<dim3((wg + 255) / 256), dim3(256), 0, stream>>>(Wk, (unsigned int*)Wkb, wg);
    cvt_bf16_kernel<<<dim3((wg + 255) / 256), dim3(256), 0, stream>>>(Wv, (unsigned int*)Wvb, wg);
  }

  dim3 g1((BATCH * T_SEQ) / 32, D_K / 128);           // (512, 6)
  qkv_proj_kernel<<<g1, dim3(256), 0, stream>>>(xbb, Wqb, Wkb, Wvb, Qb, Kb, Vtb);

  // 1024 query tiles x 2 feature halves = 2048 waves / 8 per block
  attn_kernel<<<dim3(256), dim3(256), 0, stream>>>(Qb, Kb, Vtb, (float*)d_out);
}
